// AttentionBlock_11965778886691
// MI455X (gfx1250) — matrix operand order repaired and checked by reading, not compiled, not run on hardware
//
#include <hip/hip_runtime.h>

typedef __attribute__((ext_vector_type(16))) __bf16 v16bf;
typedef __attribute__((ext_vector_type(8)))  __bf16 v8bf;
typedef __attribute__((ext_vector_type(8)))  float  v8f;
typedef int v4i_vs __attribute__((vector_size(16)));   // matches builtin's b128 pointee type

#define TILE_M 128
#define TILE_N 128
#define TILE_K 32
#define LDS_STRIDE 40   // 32 halves + 8 pad -> 80B rows: conflict-free, 16B-aligned b128 reads

#if __has_builtin(__builtin_amdgcn_global_load_async_to_lds_b128)
#define HAVE_ASYNC_COPY 1
#else
#define HAVE_ASYNC_COPY 0
#endif

__device__ __forceinline__ void async_copy_b128(void* lds, const void* g) {
#if HAVE_ASYNC_COPY
  __builtin_amdgcn_global_load_async_to_lds_b128(
      (__attribute__((address_space(1))) v4i_vs*)g,
      (__attribute__((address_space(3))) v4i_vs*)lds,
      /*offset=*/0, /*cpol=*/0);
#else
  (void)lds; (void)g;
#endif
}

__device__ __forceinline__ void wait_async0() {
#if __has_builtin(__builtin_amdgcn_s_wait_asynccnt)
  __builtin_amdgcn_s_wait_asynccnt(0);
#elif HAVE_ASYNC_COPY
  asm volatile("s_wait_asynccnt 0" ::: "memory");
#endif
}

__device__ __forceinline__ v16bf cat8(v8bf lo, v8bf hi) {
  return __builtin_shufflevector(lo, hi, 0,1,2,3,4,5,6,7,8,9,10,11,12,13,14,15);
}

// C[M,N] = A[M,K] * B' where:
//   BT=false: B is N-major (row n holds K contiguous)  -> B'[k][n] = B[n*LDB + k]
//   BT=true : B is K-major (row k holds N contiguous)  -> B'[k][n] = B[k*LDB + n]
// OUTF32: store f32, else bf16. LDA/LDB/LDC compile-time (element strides).
template<bool BT, bool OUTF32, int LDA, int LDB, int LDC>
__global__ __launch_bounds__(256)
void gemm_wmma_bf16(const __bf16* __restrict__ A, const __bf16* __restrict__ B,
                    void* __restrict__ Cv, int K,
                    long long astride, long long bstride, long long cstride)
{
  constexpr bool kAsync = (HAVE_ASYNC_COPY != 0);
  __shared__ __bf16 As[2][TILE_M * LDS_STRIDE];
  __shared__ __bf16 Bs[2][TILE_N * LDS_STRIDE];

  const int t    = threadIdx.x;
  const int lane = t & 31;
  const int wave = t >> 5;
  const int half = lane >> 4;     // K-split half per WMMA 16-bit layout
  const int r    = lane & 15;
  const int wm   = wave & 3;      // 4 waves along M (32 rows each)
  const int wn   = wave >> 2;     // 2 waves along N (64 cols each)

  const unsigned bm = blockIdx.x * TILE_M;
  const unsigned bn = blockIdx.y * TILE_N;
  A += (long long)blockIdx.z * astride;
  B += (long long)blockIdx.z * bstride;

  // staging coordinates: 512 chunks of 8 halves (16B) per tile, 2 per thread
  const int ar = t >> 2;                 // A rows ar, ar+64
  const int ac = (t & 3) << 3;
  int br, bc;
  if (!BT) { br = t >> 2;  bc = (t & 3)  << 3; }   // n-rows br, br+64
  else     { br = t >> 4;  bc = (t & 15) << 3; }   // k-rows br, br+16; n-chunk bc

  // 32-bit element offsets (max index 67.1M elements -> fits u32)
  const unsigned aoff = (bm + (unsigned)ar) * (unsigned)LDA + (unsigned)ac;
  unsigned boff;
  if (!BT) boff = (bn + (unsigned)br) * (unsigned)LDB + (unsigned)bc;
  else     boff = (unsigned)br * (unsigned)LDB + bn + (unsigned)bc;

  uint4 ra0, ra1, rb0, rb1;

  auto stage_load = [&](int kt) {
    if constexpr (!kAsync) {
      const __bf16* pa = A + aoff + (unsigned)kt * TILE_K;
      ra0 = *(const uint4*)pa;
      ra1 = *(const uint4*)(pa + 64u * (unsigned)LDA);
    }
    if constexpr (!kAsync || BT) {
      if (!BT) {
        const __bf16* pb = B + boff + (unsigned)kt * TILE_K;
        rb0 = *(const uint4*)pb;
        rb1 = *(const uint4*)(pb + 64u * (unsigned)LDB);
      } else {
        const __bf16* pb = B + boff + (unsigned)kt * TILE_K * (unsigned)LDB;
        rb0 = *(const uint4*)pb;
        rb1 = *(const uint4*)(pb + 16u * (unsigned)LDB);
      }
    }
  };

  auto stage_store = [&](int buf) {
    if constexpr (!kAsync) {
      *(uint4*)&As[buf][ar * LDS_STRIDE + ac]        = ra0;
      *(uint4*)&As[buf][(ar + 64) * LDS_STRIDE + ac] = ra1;
    }
    if (!BT) {
      if constexpr (!kAsync) {
        *(uint4*)&Bs[buf][br * LDS_STRIDE + bc]        = rb0;
        *(uint4*)&Bs[buf][(br + 64) * LDS_STRIDE + bc] = rb1;
      }
    } else {
      // transpose into Bs[n][k]
      const __bf16* e0 = (const __bf16*)&rb0;
      const __bf16* e1 = (const __bf16*)&rb1;
      #pragma unroll
      for (int e = 0; e < 8; ++e) {
        Bs[buf][(bc + e) * LDS_STRIDE + br]      = e0[e];
        Bs[buf][(bc + e) * LDS_STRIDE + br + 16] = e1[e];
      }
    }
  };

  auto async_issue = [&](int kt, int buf) {
    if constexpr (kAsync) {
      const __bf16* pa = A + aoff + (unsigned)kt * TILE_K;
      async_copy_b128(&As[buf][ar * LDS_STRIDE + ac], pa);
      async_copy_b128(&As[buf][(ar + 64) * LDS_STRIDE + ac], pa + 64u * (unsigned)LDA);
      if (!BT) {
        const __bf16* pb = B + boff + (unsigned)kt * TILE_K;
        async_copy_b128(&Bs[buf][br * LDS_STRIDE + bc], pb);
        async_copy_b128(&Bs[buf][(br + 64) * LDS_STRIDE + bc], pb + 64u * (unsigned)LDB);
      }
    }
  };

  v8f zero = {0.f,0.f,0.f,0.f,0.f,0.f,0.f,0.f};
  v8f acc[2][4];
  #pragma unroll
  for (int i = 0; i < 2; ++i)
    #pragma unroll
    for (int j = 0; j < 4; ++j) acc[i][j] = zero;

  const int nk = K >> 5;
  async_issue(0, 0);
  stage_load(0);
  stage_store(0);
  wait_async0();
  __syncthreads();

  for (int kt = 0; kt < nk; ++kt) {
    const int buf = kt & 1;
    if (kt + 1 < nk) {
      async_issue(kt + 1, buf ^ 1);
      stage_load(kt + 1);
    }
    if constexpr (!kAsync) {
      if (kt + 2 < nk) {
        __builtin_prefetch(A + aoff + (unsigned)(kt + 2) * TILE_K, 0, 1);
        if (!BT) __builtin_prefetch(B + boff + (unsigned)(kt + 2) * TILE_K, 0, 1);
        else     __builtin_prefetch(B + boff + (unsigned)(kt + 2) * TILE_K * (unsigned)LDB, 0, 1);
      }
    }

    // LDS -> fragments per ISA 16-bit WMMA lane layout (§7.12.2)
    v16bf af[2], bfr[4];
    #pragma unroll
    for (int i = 0; i < 2; ++i) {
      const __bf16* base = &As[buf][(wm * 32 + i * 16 + r) * LDS_STRIDE];
      v8bf lo = *(const v8bf*)(base + 8 * half);        // K =  8h .. 8h+7
      v8bf hi = *(const v8bf*)(base + 16 + 8 * half);   // K = 16+8h .. 16+8h+7
      af[i] = cat8(lo, hi);
    }
    #pragma unroll
    for (int j = 0; j < 4; ++j) {
      const __bf16* base = &Bs[buf][(wn * 64 + j * 16 + r) * LDS_STRIDE + 8 * half];
      v8bf lo = *(const v8bf*)(base);                   // K = 8h .. 8h+7
      v8bf hi = *(const v8bf*)(base + 16);              // K = 16+8h .. 16+8h+7
      bfr[j] = cat8(lo, hi);
    }
    #pragma unroll
    for (int i = 0; i < 2; ++i)
      #pragma unroll
      for (int j = 0; j < 4; ++j)
        acc[i][j] = __builtin_amdgcn_wmma_f32_16x16x32_bf16(
            false, af[i], false, bfr[j], (short)0, acc[i][j], false, false);

    if (kt + 1 < nk) {
      stage_store(buf ^ 1);
      wait_async0();
      __syncthreads();
    }
  }

  // epilogue: D layout -> VGPR v holds M = v + 8*half, N = lane%16
  #pragma unroll
  for (int i = 0; i < 2; ++i) {
    const unsigned row0 = bm + (unsigned)(wm * 32 + i * 16 + half * 8);
    #pragma unroll
    for (int j = 0; j < 4; ++j) {
      const unsigned col = bn + (unsigned)(wn * 64 + j * 16 + r);
      if constexpr (OUTF32) {
        float* C = (float*)Cv + (long long)blockIdx.z * cstride;
        float* p = C + (size_t)row0 * LDC + col;
        #pragma unroll
        for (int v = 0; v < 8; ++v) p[(size_t)v * LDC] = acc[i][j][v];
      } else {
        __bf16* C = (__bf16*)Cv + (long long)blockIdx.z * cstride;
        __bf16* p = C + (size_t)row0 * LDC + col;
        #pragma unroll
        for (int v = 0; v < 8; ++v) p[(size_t)v * LDC] = (__bf16)acc[i][j][v];
      }
    }
  }
}

// One block per row of 2048 f32 scores; writes 2048 bf16 probs in place over row start.
__global__ __launch_bounds__(256)
void softmax_inplace(float* __restrict__ S, float scale)
{
  __shared__ float redmax[8];
  __shared__ float redsum[8];
  const int t = threadIdx.x;
  const int lane = t & 31, wave = t >> 5;
  float* srow = S + (long long)blockIdx.x * 2048;
  __bf16* prow = (__bf16*)srow;

  float v[8];
  float m = -1e30f;
  #pragma unroll
  for (int i = 0; i < 8; ++i) {
    v[i] = srow[t + (i << 8)] * scale;
    m = fmaxf(m, v[i]);
  }
  #pragma unroll
  for (int off = 16; off > 0; off >>= 1) m = fmaxf(m, __shfl_xor(m, off, 32));
  if (lane == 0) redmax[wave] = m;
  __syncthreads();               // all reads of srow complete before any write
  float mall = redmax[0];
  #pragma unroll
  for (int w = 1; w < 8; ++w) mall = fmaxf(mall, redmax[w]);

  float p[8];
  float s = 0.f;
  #pragma unroll
  for (int i = 0; i < 8; ++i) { p[i] = __expf(v[i] - mall); s += p[i]; }
  #pragma unroll
  for (int off = 16; off > 0; off >>= 1) s += __shfl_xor(s, off, 32);
  if (lane == 0) redsum[wave] = s;
  __syncthreads();
  float sall = 0.f;
  #pragma unroll
  for (int w = 0; w < 8; ++w) sall += redsum[w];
  const float inv = 1.f / sall;

  #pragma unroll
  for (int i = 0; i < 8; ++i) prow[t + (i << 8)] = (__bf16)(p[i] * inv);
}

__global__ void cvt_f32_to_bf16(const float* __restrict__ in, __bf16* __restrict__ out, int n)
{
  int i = blockIdx.x * blockDim.x + threadIdx.x;
  if (i < n) out[i] = (__bf16)in[i];
}

__global__ void pack_weights_bf16(const float* __restrict__ Wq, const float* __restrict__ Wk,
                                  const float* __restrict__ Wv, __bf16* __restrict__ Wall)
{
  int i = blockIdx.x * blockDim.x + threadIdx.x;        // 0 .. 1536*512-1
  int row = i >> 9;
  const float* src = (row < 512) ? Wq : (row < 1024 ? Wk : Wv);
  int j = (i & 511) + ((row & 511) << 9);
  Wall[i] = (__bf16)src[j];
}

extern "C" void kernel_launch(void* const* d_in, const int* in_sizes, int n_in,
                              void* d_out, int out_size, void* d_ws, size_t ws_size,
                              hipStream_t stream)
{
  const float* x  = (const float*)d_in[0];
  const float* Wq = (const float*)d_in[1];
  const float* Wk = (const float*)d_in[2];
  const float* Wv = (const float*)d_in[3];
  float* out = (float*)d_out;

  const int Bb = 8, S = 2048;
  const long long rows = (long long)Bb * S;             // 16384

  // workspace layout (bytes)
  const size_t off_W   = 16777216;                      // xb: 16384*512*2
  const size_t off_qkv = 18350080;                      // Wall: 1536*512*2
  const size_t off_S   = 68681728;                      // QKV: 16384*1536*2
  const size_t need_all = off_S + (size_t)Bb * S * S * 4;   // ~203 MB
  char* ws = (char*)d_ws;
  __bf16* xb   = (__bf16*)(ws);
  __bf16* Wall = (__bf16*)(ws + off_W);
  __bf16* QKV  = (__bf16*)(ws + off_qkv);
  float*  Sb   = (float*) (ws + off_S);

  const float scale = 0.044194173824159216f;            // 1/sqrt(512)

  cvt_f32_to_bf16<<<dim3(32768), dim3(256), 0, stream>>>(x, xb, (int)(rows * 512));
  pack_weights_bf16<<<dim3(3072), dim3(256), 0, stream>>>(Wq, Wk, Wv, Wall);

  // QKV[16384 x 1536] = xb[16384 x 512] * Wall^T  (B N-major, bf16 out)
  gemm_wmma_bf16<false, false, 512, 512, 1536><<<dim3(128, 12, 1), dim3(256), 0, stream>>>(
      xb, Wall, (void*)QKV, 512, 0, 0, 0);

  if (ws_size >= need_all) {
    // scores[b] = Q_b * K_b^T  (f32 out, batched over z)
    gemm_wmma_bf16<false, true, 1536, 1536, 2048><<<dim3(16, 16, 8), dim3(256), 0, stream>>>(
        QKV, QKV + 512, (void*)Sb, 512,
        (long long)S * 1536, (long long)S * 1536, (long long)S * 2048);

    softmax_inplace<<<dim3((unsigned)rows), dim3(256), 0, stream>>>(Sb, scale);

    // out[b] = P_b * V_b  (P bf16 in-place over scores, lda in halves = 4096; V K-major)
    gemm_wmma_bf16<true, true, 4096, 1536, 512><<<dim3(16, 4, 8), dim3(256), 0, stream>>>(
        (const __bf16*)Sb, QKV + 1024, (void*)out, 2048,
        (long long)S * 4096, (long long)S * 1536, (long long)S * 512);
  } else {
    // small-workspace fallback: reuse one 2048x2048 score buffer per batch (same stream -> serialized)
    for (int b = 0; b < Bb; ++b) {
      const __bf16* Qb = QKV + (long long)b * S * 1536;
      gemm_wmma_bf16<false, true, 1536, 1536, 2048><<<dim3(16, 16, 1), dim3(256), 0, stream>>>(
          Qb, Qb + 512, (void*)Sb, 512, 0, 0, 0);
      softmax_inplace<<<dim3(S), dim3(256), 0, stream>>>(Sb, scale);
      gemm_wmma_bf16<true, true, 4096, 1536, 512><<<dim3(16, 4, 1), dim3(256), 0, stream>>>(
          (const __bf16*)Sb, Qb + 1024, (void*)(out + (long long)b * S * 512),
          2048, 0, 0, 0);
    }
  }
}
